// RPN_58308476010925
// MI455X (gfx1250) — matrix operand order repaired and checked by reading, not compiled, not run on hardware
//
#include <hip/hip_runtime.h>
#include <math.h>

// ---------------------------------------------------------------------------
// RPN head for MI455X (gfx1250, wave32, WMMA).
// Conv 3x3 (256->256) + ReLU as implicit GEMM with v_wmma_f32_16x16x32_bf16
// (64oc x 16pos macro-tile per wave, 4 accumulators sharing B; all A-fragment
// loads issued as one burst so WMMAs overlap outstanding loads),
// fused 1x1 heads as a second WMMA GEMM, then top-k / NMS / box decode.
// ---------------------------------------------------------------------------

typedef __attribute__((ext_vector_type(16))) __bf16 v16bf;
typedef __attribute__((ext_vector_type(8)))  float  v8f;

struct __align__(16) BF8 { __bf16 v[8]; };

#define NBATCH   4
#define NCH      256
#define NPOS     5440          // 4096 + 1024 + 256 + 64
#define NANCH    16320         // NPOS * 3
#define PRE_K    2000
#define POST_K   1000
#define SORT_N   2048

// ---------------- conversion kernels ----------------

__global__ __launch_bounds__(256) void cvt_feats_kernel(
    const float* __restrict__ f0, const float* __restrict__ f1,
    const float* __restrict__ f2, const float* __restrict__ f3,
    __bf16* __restrict__ out) {
  int idx = blockIdx.x * 256 + threadIdx.x;          // total = 4*5440*256
  if (idx >= NBATCH * NPOS * NCH) return;
  int c    = idx & 255;
  int rest = idx >> 8;
  int pos  = rest % NPOS;
  int b    = rest / NPOS;
  const float* src; int lpos, HW;
  if (pos < 4096)      { src = f0; lpos = pos;        HW = 64; }
  else if (pos < 5120) { src = f1; lpos = pos - 4096; HW = 32; }
  else if (pos < 5376) { src = f2; lpos = pos - 5120; HW = 16; }
  else                 { src = f3; lpos = pos - 5376; HW = 8;  }
  float v = src[((b * NCH + c) * HW * HW) + lpos];   // NCHW source
  out[idx] = (__bf16)v;                               // NHWC bf16, levels concat
}

__global__ __launch_bounds__(256) void cvt_weights_kernel(
    const float* __restrict__ w_conv, const float* __restrict__ w_cls,
    const float* __restrict__ w_box,
    __bf16* __restrict__ wt, __bf16* __restrict__ wcb) {
  int idx = blockIdx.x * 256 + threadIdx.x;
  const int NW = 9 * NCH * NCH;
  if (idx < NW) {
    int ic = idx & 255; int r = idx >> 8; int oc = r & 255; int tap = r >> 8;
    // w_conv layout (oc, ic, ky, kx): (oc*256+ic)*9 + tap
    wt[idx] = (__bf16)w_conv[(oc * NCH + ic) * 9 + tap];
  } else if (idx < NW + 16 * NCH) {
    int j = idx - NW; int ic = j & 255; int row = j >> 8;
    float v = 0.0f;
    if (row < 3)       v = w_cls[row * NCH + ic];
    else if (row < 15) v = w_box[(row - 3) * NCH + ic];
    wcb[j] = (__bf16)v;
  }
}

// ---------------- conv 3x3 + ReLU (implicit GEMM, WMMA) ----------------
// one wave per 64(oc) x 16(pos) macro-tile; 4 accumulators share each B
// fragment -> 40 B/lane of loads per WMMA and 4 independent XDL chains.

__global__ __launch_bounds__(32) void conv3x3_wmma_kernel(
    const __bf16* __restrict__ feats, const __bf16* __restrict__ wt,
    const float* __restrict__ b_conv, __bf16* __restrict__ tout) {
  const int lvlTileBase[4] = {0, 256, 320, 336};      // spatial-tile bases
  const int lvlPosBase[4]  = {0, 4096, 5120, 5376};
  const int lvlHW[4]       = {64, 32, 16, 8};

  int id   = blockIdx.x;                  // b * (4*340) + ocQ*340 + st
  int b    = id / (4 * 340);
  int r    = id % (4 * 340);
  int ocQ  = r / 340;                     // which 64-channel output group
  int st   = r % 340;
  int lvl  = (st < 256) ? 0 : (st < 320 ? 1 : (st < 336 ? 2 : 3));
  int HW   = lvlHW[lvl];
  int posBase = (st - lvlTileBase[lvl]) * 16;

  int L  = threadIdx.x;
  int Nl = L & 15;        // A row (oc within tile) AND D column (pos)
  int hh = L >> 4;        // K half selector

  int pos = posBase + Nl;
  int y = pos / HW, x = pos % HW;
  const __bf16* fb = feats + (long long)(b * NPOS + lvlPosBase[lvl]) * NCH;
  int ocBase = ocQ * 64;

  v8f acc[4] = {{}, {}, {}, {}};
  for (int tap = 0; tap < 9; ++tap) {
    int dy = tap / 3 - 1, dx = tap % 3 - 1;
    int sy = y + dy, sx = x + dx;
    bool valid = (sy >= 0) & (sy < HW) & (sx >= 0) & (sx < HW);
    const __bf16* brow = fb + (long long)(sy * HW + sx) * NCH + hh * 8;
    const __bf16* arow = wt + (tap * NCH + ocBase + Nl) * NCH + hh * 8;
    for (int kc = 0; kc < NCH; kc += 32) {
      // issue ALL fragment loads first (one clause burst), WMMAs after ->
      // backend can retire WMMA t while loads for t+1.. are still in flight
      BF8 a0[4], a1[4];
      #pragma unroll
      for (int t = 0; t < 4; ++t) {
        const BF8* ap = (const BF8*)(arow + t * 16 * NCH + kc);
        a0[t] = ap[0];
        a1[t] = ap[2];
      }
      // B fragment (32x16 bf16): elems 0..7 -> K = kc + hh*8 + e, 8..15 -> +16
      v16bf bv;
      if (valid) {
        const BF8* bp = (const BF8*)(brow + kc);
        BF8 b0 = bp[0], b1 = bp[2];
        #pragma unroll
        for (int e = 0; e < 8; ++e) { bv[e] = b0.v[e]; bv[e + 8] = b1.v[e]; }
        __builtin_prefetch(brow + kc + 32, 0, 1);   // global_prefetch path
      } else {
        #pragma unroll
        for (int e = 0; e < 16; ++e) bv[e] = (__bf16)0.0f;
      }
      // 4 oc-tiles reuse bv; A fragment elems 0..7 -> K=kc+hh*8+0..7, 8..15 -> +16
      #pragma unroll
      for (int t = 0; t < 4; ++t) {
        v16bf av;
        #pragma unroll
        for (int e = 0; e < 8; ++e) { av[e] = a0[t].v[e]; av[e + 8] = a1[t].v[e]; }
        acc[t] = __builtin_amdgcn_wmma_f32_16x16x32_bf16(
            false, av, false, bv, (short)0, acc[t], false, false);
      }
    }
  }
  // epilogue: D elem v -> M = v + hh*8 (oc), N = Nl (pos); bias + ReLU,
  // packed 16-byte store per oc-tile into NHWC bf16
  __bf16* tb = tout + (long long)(b * NPOS + lvlPosBase[lvl] + pos) * NCH;
  #pragma unroll
  for (int t = 0; t < 4; ++t) {
    BF8 ov;
    #pragma unroll
    for (int v = 0; v < 8; ++v) {
      int m = v + hh * 8;
      float val = acc[t][v] + b_conv[ocBase + t * 16 + m];
      ov.v[v] = (__bf16)(val > 0.0f ? val : 0.0f);
    }
    *(BF8*)(tb + ocBase + t * 16 + hh * 8) = ov;
  }
}

// ---------------- fused 1x1 heads (cls sigmoid + box deltas) ----------------

__global__ __launch_bounds__(32) void head_wmma_kernel(
    const __bf16* __restrict__ tfeat, const __bf16* __restrict__ wcb,
    const float* __restrict__ b_cls, const float* __restrict__ b_box,
    float* __restrict__ scores, float* __restrict__ deltas) {
  const int lvlTileBase[4] = {0, 256, 320, 336};
  const int lvlPosBase[4]  = {0, 4096, 5120, 5376};
  const int aBase[4]       = {0, 12288, 15360, 16128};

  int id  = blockIdx.x;                 // b*340 + st
  int b   = id / 340;
  int st  = id % 340;
  int lvl = (st < 256) ? 0 : (st < 320 ? 1 : (st < 336 ? 2 : 3));
  int posBase = (st - lvlTileBase[lvl]) * 16;

  int L = threadIdx.x; int Nl = L & 15; int hh = L >> 4;
  const __bf16* tb   = tfeat + (long long)(b * NPOS + lvlPosBase[lvl] + posBase) * NCH;
  const __bf16* arow = wcb + Nl * NCH + hh * 8;
  const __bf16* brow = tb  + Nl * NCH + hh * 8;

  v8f acc = {};
  for (int kc = 0; kc < NCH; kc += 32) {
    const BF8* ap = (const BF8*)(arow + kc);
    const BF8* bp = (const BF8*)(brow + kc);
    BF8 a0 = ap[0], a1 = ap[2];
    BF8 b0 = bp[0], b1 = bp[2];
    v16bf av, bv;
    #pragma unroll
    for (int e = 0; e < 8; ++e) {
      av[e] = a0.v[e]; av[e + 8] = a1.v[e];
      bv[e] = b0.v[e]; bv[e + 8] = b1.v[e];
    }
    acc = __builtin_amdgcn_wmma_f32_16x16x32_bf16(
        false, av, false, bv, (short)0, acc, false, false);
  }
  int pos   = posBase + Nl;
  int abase = aBase[lvl] + pos * 3;
  #pragma unroll
  for (int v = 0; v < 8; ++v) {
    int m = v + hh * 8;                 // head output channel
    float val = acc[v];
    if (m < 3) {                        // cls -> sigmoid
      float s = val + b_cls[m];
      s = 1.0f / (1.0f + __expf(-s));
      scores[b * NANCH + abase + m] = s;
    } else if (m < 15) {                // box deltas
      int j = m - 3; int k = j >> 2; int c = j & 3;
      deltas[(long long)(b * NANCH + abase + k) * 4 + c] = val + b_box[j];
    }
  }
}

// ---------------- exact deterministic top-2000 ----------------

__device__ inline unsigned long long make_key(float s, int i) {
  unsigned int u = __float_as_uint(s);
  u = (u & 0x80000000u) ? ~u : (u | 0x80000000u);       // order-preserving
  return (((unsigned long long)u) << 32) | (unsigned int)(0xFFFFFFFFu - (unsigned)i);
}

__global__ __launch_bounds__(256) void topk_pre_kernel(
    const float* __restrict__ scores, int* __restrict__ topIdx) {
  int b = blockIdx.x, tid = threadIdx.x;
  const float* sc = scores + b * NANCH;
  __shared__ int red[256];
  __shared__ unsigned long long skey[SORT_N];
  __shared__ int cnt;

  // binary search for the PRE_K-th largest unique 64-bit key
  unsigned long long lo = 0ULL, hi = ~0ULL;
  for (int it = 0; it < 64; ++it) {
    if (lo >= hi) break;
    unsigned long long mid = lo + ((hi - lo) >> 1) + ((hi - lo) & 1ULL);
    int c = 0;
    for (int i = tid; i < NANCH; i += 256)
      c += (make_key(sc[i], i) >= mid) ? 1 : 0;
    red[tid] = c; __syncthreads();
    for (int s = 128; s > 0; s >>= 1) {
      if (tid < s) red[tid] += red[tid + s];
      __syncthreads();
    }
    int total = red[0];
    __syncthreads();
    if (total >= PRE_K) lo = mid; else hi = mid - 1;
  }
  // select + LDS bitonic sort (descending)
  if (tid == 0) cnt = 0;
  for (int i = tid; i < SORT_N; i += 256) skey[i] = 0ULL;
  __syncthreads();
  for (int i = tid; i < NANCH; i += 256) {
    unsigned long long k = make_key(sc[i], i);
    if (k >= lo) { int p = atomicAdd(&cnt, 1); if (p < SORT_N) skey[p] = k; }
  }
  __syncthreads();
  for (int ksz = 2; ksz <= SORT_N; ksz <<= 1) {
    for (int jj = ksz >> 1; jj > 0; jj >>= 1) {
      for (int t = tid; t < SORT_N; t += 256) {
        int ixj = t ^ jj;
        if (ixj > t) {
          unsigned long long a = skey[t], c = skey[ixj];
          bool desc = ((t & ksz) == 0);
          if (desc ? (a < c) : (a > c)) { skey[t] = c; skey[ixj] = a; }
        }
      }
      __syncthreads();
    }
  }
  for (int j = tid; j < SORT_N; j += 256) {
    unsigned long long k = skey[j];
    int i = (k == 0ULL) ? 0 : (int)(0xFFFFFFFFu - (unsigned)(k & 0xFFFFFFFFu));
    topIdx[b * SORT_N + j] = i;
  }
}

// ---------------- anchors (analytic) + NMS ----------------

__device__ inline void decode_anchor(int a, float* cx, float* cy, float* w, float* h) {
  int lvl = (a < 12288) ? 0 : (a < 15360 ? 1 : (a < 16128 ? 2 : 3));
  const int   aB[4]   = {0, 12288, 15360, 16128};
  const int   HWs[4]  = {64, 32, 16, 8};
  const float strd[4] = {8.f, 16.f, 32.f, 64.f};
  const float szs[4]  = {64.f, 128.f, 256.f, 512.f};
  const float sqr[3]  = {0.70710678118654752f, 1.0f, 1.41421356237309515f};
  int rel = a - aB[lvl];
  int pos = rel / 3, k = rel - pos * 3;
  int Wd = HWs[lvl];
  int yy = pos / Wd, xx = pos - yy * Wd;
  *cx = (xx + 0.5f) * strd[lvl];
  *cy = (yy + 0.5f) * strd[lvl];
  *w  = szs[lvl] / sqr[k];
  *h  = szs[lvl] * sqr[k];
}

__global__ __launch_bounds__(256) void nms_kernel(
    const int* __restrict__ topIdx, int* __restrict__ keep) {
  int b = blockIdx.x, tid = threadIdx.x;
  __shared__ float bx[PRE_K][4];
  __shared__ unsigned char kp[PRE_K];
  for (int i = tid; i < PRE_K; i += 256) {
    float cx, cy, w, h;
    decode_anchor(topIdx[b * SORT_N + i], &cx, &cy, &w, &h);
    bx[i][0] = cx - w * 0.5f; bx[i][1] = cy - h * 0.5f;
    bx[i][2] = cx + w * 0.5f; bx[i][3] = cy + h * 0.5f;
    kp[i] = 1;
  }
  __syncthreads();
  for (int i = 0; i < PRE_K; ++i) {
    if (kp[i]) {
      float x1 = bx[i][0], y1 = bx[i][1], x2 = bx[i][2], y2 = bx[i][3];
      float ai = (x2 - x1) * (y2 - y1);
      for (int j = i + 1 + tid; j < PRE_K; j += 256) {
        if (kp[j]) {
          float xx1 = fmaxf(x1, bx[j][0]), yy1 = fmaxf(y1, bx[j][1]);
          float xx2 = fminf(x2, bx[j][2]), yy2 = fminf(y2, bx[j][3]);
          float iw = fmaxf(xx2 - xx1, 0.f), ih = fmaxf(yy2 - yy1, 0.f);
          float inter = iw * ih;
          float aj = (bx[j][2] - bx[j][0]) * (bx[j][3] - bx[j][1]);
          if (inter / (ai + aj - inter + 1e-9f) > 0.7f) kp[j] = 0;
        }
      }
    }
    __syncthreads();
  }
  for (int i = tid; i < PRE_K; i += 256) keep[b * SORT_N + i] = kp[i];
}

// ---------------- post-NMS top-1000 + box decode ----------------

__global__ __launch_bounds__(256) void final_kernel(
    const int* __restrict__ topIdx, const int* __restrict__ keep,
    const float* __restrict__ deltas, const int* __restrict__ img_size,
    float* __restrict__ out) {
  int b = blockIdx.x, tid = threadIdx.x;
  __shared__ int pi[POST_K];
  if (tid == 0) {   // stable: kept (sorted desc) first, then suppressed by index
    int c = 0;
    for (int i = 0; i < PRE_K && c < POST_K; ++i) if (keep[b * SORT_N + i])  pi[c++] = i;
    for (int i = 0; i < PRE_K && c < POST_K; ++i) if (!keep[b * SORT_N + i]) pi[c++] = i;
  }
  __syncthreads();
  float ihg = (float)img_size[b * 2 + 0];
  float iwg = (float)img_size[b * 2 + 1];
  for (int j = tid; j < POST_K; j += 256) {
    int i = pi[j];
    int a = topIdx[b * SORT_N + i];
    float cx, cy, w, h;
    decode_anchor(a, &cx, &cy, &w, &h);
    const float* d = deltas + (long long)(b * NANCH + a) * 4;
    float d2 = fminf(fmaxf(d[2], -4.135f), 4.135f);
    float d3 = fminf(fmaxf(d[3], -4.135f), 4.135f);
    float ncx = cx + d[0] * w, ncy = cy + d[1] * h;
    float nw = w * __expf(d2), nh = h * __expf(d3);
    float x1 = fminf(fmaxf(ncx - nw * 0.5f, 0.f), iwg);
    float x2 = fminf(fmaxf(ncx + nw * 0.5f, 0.f), iwg);
    float y1 = fminf(fmaxf(ncy - nh * 0.5f, 0.f), ihg);
    float y2 = fminf(fmaxf(ncy + nh * 0.5f, 0.f), ihg);
    float* o = out + (long long)(b * POST_K + j) * 4;
    o[0] = (x1 + x2) * 0.5f; o[1] = (y1 + y2) * 0.5f;
    o[2] = x2 - x1;          o[3] = y2 - y1;
  }
}

// ---------------- launcher ----------------

extern "C" void kernel_launch(void* const* d_in, const int* in_sizes, int n_in,
                              void* d_out, int out_size, void* d_ws, size_t ws_size,
                              hipStream_t stream) {
  (void)in_sizes; (void)n_in; (void)out_size; (void)ws_size;
  const float* f0     = (const float*)d_in[0];
  const float* f1     = (const float*)d_in[1];
  const float* f2     = (const float*)d_in[2];
  const float* f3     = (const float*)d_in[3];
  const int*   img    = (const int*)d_in[4];
  const float* w_conv = (const float*)d_in[5];
  const float* b_conv = (const float*)d_in[6];
  const float* w_cls  = (const float*)d_in[7];
  const float* b_cls  = (const float*)d_in[8];
  const float* w_box  = (const float*)d_in[9];
  const float* b_box  = (const float*)d_in[10];

  char* ws = (char*)d_ws;
  // workspace layout (256B aligned offsets)
  __bf16* featsBF = (__bf16*)(ws + 0);          // 4*5440*256*2  = 11,141,120
  __bf16* tBF     = (__bf16*)(ws + 11141120);   // 11,141,120
  __bf16* wt      = (__bf16*)(ws + 22282240);   // 9*256*256*2   =  1,179,648
  __bf16* wcb     = (__bf16*)(ws + 23461888);   // 16*256*2      =      8,192
  float*  scores  = (float*) (ws + 23470080);   // 4*16320*4     =    261,120
  float*  deltas  = (float*) (ws + 23731200);   // 4*16320*4*4   =  1,044,480
  int*    topIdx  = (int*)   (ws + 24775680);   // 4*2048*4      =     32,768
  int*    keep    = (int*)   (ws + 24808448);   // 4*2048*4      =     32,768

  // 1) fp32 NCHW -> bf16 NHWC (levels concatenated)
  cvt_feats_kernel<<<(NBATCH * NPOS * NCH) / 256, 256, 0, stream>>>(f0, f1, f2, f3, featsBF);
  // 2) weights -> bf16, [tap][oc][ic] and fused head [16][256]
  cvt_weights_kernel<<<(9 * NCH * NCH + 16 * NCH) / 256, 256, 0, stream>>>(
      w_conv, w_cls, w_box, wt, wcb);
  // 3) conv3x3 + bias + ReLU via WMMA: 4 batches * 4 oc-quads * 340 pos-tiles
  conv3x3_wmma_kernel<<<NBATCH * 4 * 340, 32, 0, stream>>>(featsBF, wt, b_conv, tBF);
  // 4) fused 1x1 heads via WMMA
  head_wmma_kernel<<<NBATCH * 340, 32, 0, stream>>>(tBF, wcb, b_cls, b_box, scores, deltas);
  // 5) exact top-2000 per batch
  topk_pre_kernel<<<NBATCH, 256, 0, stream>>>(scores, topIdx);
  // 6) greedy NMS on anchor boxes
  nms_kernel<<<NBATCH, 256, 0, stream>>>(topIdx, keep);
  // 7) stable top-1000 + box regression + clip
  final_kernel<<<NBATCH, 256, 0, stream>>>(topIdx, keep, deltas, img, (float*)d_out);
}
